// BatchWiseTripletLoss_13786845020972
// MI455X (gfx1250) — matrix operand order repaired and checked by reading, not compiled, not run on hardware
//
#include <hip/hip_runtime.h>

// Batch-wise triplet loss, n=8192, d=128, MI455X (gfx1250, wave32).
// bf16 WMMA (fp32 accum) over L2-resident normalized embeddings; two fused
// GEMM passes per workgroup-owned 32-row strip. A fragments live in VGPRs
// (loaded once from global), B tiles double-buffered in LDS via
// GLOBAL_LOAD_ASYNC_TO_LDS_B128. Branchless select epilogues; select-based
// max (no fmaxf canonicalize) to minimize VALU issue under the WMMAs.

#define NROWS 8192
#define DIM   128
#define TM    32          // rows owned per workgroup
#define TN    64          // columns per LDS tile iteration
#define LDK   136         // 128 + 8 bf16 pad (16B) -> stagger LDS banks
#define MARGIN_F 0.1f
#define NEG_FLOOR_F 0.6f
#define NEG_INF_F (-3.4e38f)

typedef __attribute__((ext_vector_type(16))) __bf16 v16bf;
typedef __attribute__((ext_vector_type(8)))  float  v8f;

union Frag {
  v16bf v;
  uint4 q[2];
};

#if __has_builtin(__builtin_amdgcn_global_load_async_to_lds_b128) && \
    __has_builtin(__builtin_amdgcn_s_wait_asynccnt)
#define USE_ASYNC_LDS 1
#define WAIT_ASYNC(n) __builtin_amdgcn_s_wait_asynccnt(n)
// Exact parameter types per hipcc diagnostic: pointer to 16B int vector in
// AS1 (global / "__device__") and AS3 (LDS / "__shared__").
typedef int v4i_b128 __attribute__((vector_size(16)));
typedef __attribute__((address_space(1))) v4i_b128* gas_b128_t;
typedef __attribute__((address_space(3))) v4i_b128* las_b128_t;
#endif

// Finite-input max: v_cmp + v_cndmask, avoids fmaxf's canonicalize ops.
__device__ __forceinline__ float selmax(float a, float b) {
  return a > b ? a : b;
}

__device__ __forceinline__ unsigned short f2bf_rne(float f) {
  unsigned int u = __float_as_uint(f);
  u += 0x7fffu + ((u >> 16) & 1u);   // round-to-nearest-even
  return (unsigned short)(u >> 16);
}

// One wave per row: L2-normalize, convert to bf16, store to workspace.
__global__ __launch_bounds__(256) void normalize_rows_bf16(
    const float* __restrict__ emb, unsigned short* __restrict__ e16,
    float* __restrict__ out) {
  if (blockIdx.x == 0 && threadIdx.x == 0) out[0] = 0.0f;  // init accumulator
  const int wave = threadIdx.x >> 5;
  const int lane = threadIdx.x & 31;
  const int row  = blockIdx.x * 8 + wave;
  const float4 v = reinterpret_cast<const float4*>(emb + (size_t)row * DIM)[lane];
  float ss = v.x * v.x + v.y * v.y + v.z * v.z + v.w * v.w;
#pragma unroll
  for (int off = 1; off < 32; off <<= 1) ss += __shfl_xor(ss, off, 32);
  const float scale = 1.0f / fmaxf(sqrtf(ss), 1e-12f);
  ushort4 o;
  o.x = f2bf_rne(v.x * scale);
  o.y = f2bf_rne(v.y * scale);
  o.z = f2bf_rne(v.z * scale);
  o.w = f2bf_rne(v.w * scale);
  reinterpret_cast<ushort4*>(e16 + (size_t)row * DIM)[lane] = o;
}

// Stage one 64-row B tile (16KB) + its 64 class labels into LDS.
__device__ __forceinline__ void stage_b_tile(
    const unsigned short* __restrict__ e16, const int* __restrict__ tgt,
    int jbase, unsigned short* lBbuf, int* tColBuf, int tid) {
#pragma unroll
  for (int t = 0; t < 4; ++t) {
    int chunk = tid + t * 256;          // 1024 x 16B chunks
    int row = chunk >> 4;
    int c8  = chunk & 15;
    const unsigned short* src = e16 + (size_t)(jbase + row) * DIM + (size_t)c8 * 8;
    unsigned short* dst = lBbuf + row * LDK + c8 * 8;
#if defined(USE_ASYNC_LDS)
    __builtin_amdgcn_global_load_async_to_lds_b128(
        (gas_b128_t)const_cast<unsigned short*>(src), (las_b128_t)dst, 0, 0);
#else
    *reinterpret_cast<uint4*>(dst) = *reinterpret_cast<const uint4*>(src);
#endif
  }
  if (tid < TN) tColBuf[tid] = tgt[jbase + tid];
}

// 16x16 tile GEMM over K=128: A fragments resident in VGPRs, B from LDS.
__device__ __forceinline__ v8f tile_gemm(const Frag* afr, const uint4* lB4,
                                         int bbase) {
  v8f c = {};
#pragma unroll
  for (int k = 0; k < 4; ++k) {
    Frag b;
    int boff = (bbase + k * 32) >> 3;
    b.q[0] = lB4[boff];
    b.q[1] = lB4[boff + 1];
    c = __builtin_amdgcn_wmma_f32_16x16x32_bf16(false, afr[k].v, false, b.v,
                                                (short)0, c, false, false);
  }
  return c;
}

__global__ __launch_bounds__(256) void triplet_wmma(
    const unsigned short* __restrict__ e16, const int* __restrict__ tgt,
    float* __restrict__ out) {
  __shared__ __align__(16) unsigned short lB[2][TN * LDK];
  __shared__ int   tColB[2][TN];
  __shared__ float sNeg[8][16];
  __shared__ float sPos[8][16];
  __shared__ float sLoss[8][16];
  __shared__ float rowMaxNeg[TM];
  __shared__ float rowMaxPos[TM];

  const int tid    = threadIdx.x;
  const int wave   = tid >> 5;
  const int lane   = tid & 31;
  const int ibase  = blockIdx.x * TM;
  const int mBlock = wave & 1;        // 2 row-blocks of 16
  const int nBlock = wave >> 1;       // 4 col-blocks of 16
  const int nIdx   = lane & 15;       // C column / B-fragment N / A-fragment M
  const int half8  = (lane >> 4) << 3;        // C rows 0..7 vs 8..15
  const int kbA    = (lane < 16) ? 0 : 16;    // A-frag K base (contiguous, matches B)
  const int kbB    = (lane < 16) ? 0 : 16;    // B-frag K base (contiguous, matches A)
  const int NT     = NROWS / TN;

  // A fragments: one row per lane, entirely register-resident for the kernel.
  Frag afr[4];
  {
    const unsigned short* arow =
        e16 + (size_t)(ibase + mBlock * 16 + nIdx) * DIM;
#pragma unroll
    for (int k = 0; k < 4; ++k) {
      afr[k].q[0] = *reinterpret_cast<const uint4*>(arow + k * 32 + kbA);
      afr[k].q[1] = *reinterpret_cast<const uint4*>(arow + k * 32 + kbA + 8);
    }
  }
  // Class labels of this wave's 8 C rows (same for all 16 lanes of a half).
  int trowv[8];
#pragma unroll
  for (int r = 0; r < 8; ++r)
    trowv[r] = tgt[ibase + mBlock * 16 + half8 + r];

  const int bbase = (nBlock * 16 + nIdx) * LDK + kbB;

  // ---------------- Pass 1: hardest negative / positive per row ----------
  float mxN[8], mxP[8];
#pragma unroll
  for (int r = 0; r < 8; ++r) { mxN[r] = NEG_INF_F; mxP[r] = NEG_INF_F; }

  stage_b_tile(e16, tgt, 0, lB[0], tColB[0], tid);
  for (int it = 0; it < NT; ++it) {
    const int cur = it & 1;
    if (it + 1 < NT) {
      stage_b_tile(e16, tgt, (it + 1) * TN, lB[cur ^ 1], tColB[cur ^ 1], tid);
#if defined(USE_ASYNC_LDS)
      WAIT_ASYNC(4);   // in-order: current tile's 4 async loads complete
#endif
    } else {
#if defined(USE_ASYNC_LDS)
      WAIT_ASYNC(0);
#endif
    }
    __syncthreads();   // current tile visible to all waves

    v8f c = tile_gemm(afr, reinterpret_cast<const uint4*>(lB[cur]), bbase);

    const int jbase = it * TN;
    const int jcol  = jbase + nBlock * 16 + nIdx;
    const int tcls  = tColB[cur][nBlock * 16 + nIdx];
#pragma unroll
    for (int r = 0; r < 8; ++r) {
      float s = c[r];
      int irow  = ibase + mBlock * 16 + half8 + r;
      bool same = (tcls == trowv[r]);
      float candN = same ? NEG_INF_F : s;                      // negatives
      float candP = (same && irow != jcol) ? s : NEG_INF_F;    // positives
      mxN[r] = selmax(mxN[r], candN);
      mxP[r] = selmax(mxP[r], candP);
    }
    __syncthreads();   // all waves done with buf[cur] before it is re-staged
  }

  // Reduce row maxima across the 16 lanes holding each C row, then across
  // the 4 waves that share each 16-row block.
#pragma unroll
  for (int r = 0; r < 8; ++r) {
#pragma unroll
    for (int off = 1; off <= 8; off <<= 1) {
      mxN[r] = selmax(mxN[r], __shfl_xor(mxN[r], off, 32));
      mxP[r] = selmax(mxP[r], __shfl_xor(mxP[r], off, 32));
    }
  }
  if ((lane & 15) == 0) {
#pragma unroll
    for (int r = 0; r < 8; ++r) {
      sNeg[wave][half8 + r] = mxN[r];
      sPos[wave][half8 + r] = mxP[r];
    }
  }
  __syncthreads();
  if (tid < TM) {
    float mn = NEG_INF_F, mp = NEG_INF_F;
    for (int w = (tid >> 4); w < 8; w += 2) {
      mn = selmax(mn, sNeg[w][tid & 15]);
      mp = selmax(mp, sPos[w][tid & 15]);
    }
    rowMaxNeg[tid] = mn;
    rowMaxPos[tid] = mp;
  }
  __syncthreads();

  // ---------------- Pass 2: thresholded hard-example sums ----------------
  float thrP[8], thrN[8];
#pragma unroll
  for (int r = 0; r < 8; ++r) {
    int rl = mBlock * 16 + half8 + r;
    thrP[r] = rowMaxNeg[rl] + MARGIN_F;
    thrN[r] = selmax(NEG_FLOOR_F, rowMaxPos[rl]) - MARGIN_F;
  }
  float acc[8];
#pragma unroll
  for (int r = 0; r < 8; ++r) acc[r] = 0.0f;

  stage_b_tile(e16, tgt, 0, lB[0], tColB[0], tid);
  for (int it = 0; it < NT; ++it) {
    const int cur = it & 1;
    if (it + 1 < NT) {
      stage_b_tile(e16, tgt, (it + 1) * TN, lB[cur ^ 1], tColB[cur ^ 1], tid);
#if defined(USE_ASYNC_LDS)
      WAIT_ASYNC(4);
#endif
    } else {
#if defined(USE_ASYNC_LDS)
      WAIT_ASYNC(0);
#endif
    }
    __syncthreads();

    v8f c = tile_gemm(afr, reinterpret_cast<const uint4*>(lB[cur]), bbase);

    const int jbase = it * TN;
    const int jcol  = jbase + nBlock * 16 + nIdx;
    const int tcls  = tColB[cur][nBlock * 16 + nIdx];
#pragma unroll
    for (int r = 0; r < 8; ++r) {
      float s = c[r];
      int irow  = ibase + mBlock * 16 + half8 + r;
      bool same = (tcls == trowv[r]);
      float posTerm = (same && irow != jcol && s < thrP[r]) ? (1.0f - s) : 0.0f;
      float negTerm = (!same && s > thrN[r]) ? s : 0.0f;
      acc[r] += posTerm + negTerm;
    }
    __syncthreads();
  }

  // Sum per row across lanes, combine across waves, gate by has_pos, reduce.
#pragma unroll
  for (int r = 0; r < 8; ++r) {
#pragma unroll
    for (int off = 1; off <= 8; off <<= 1)
      acc[r] += __shfl_xor(acc[r], off, 32);
  }
  if ((lane & 15) == 0) {
#pragma unroll
    for (int r = 0; r < 8; ++r) sLoss[wave][half8 + r] = acc[r];
  }
  __syncthreads();
  if (tid < TM) {
    float loss = 0.0f;
    for (int w = (tid >> 4); w < 8; w += 2) loss += sLoss[w][tid & 15];
    if (!(rowMaxPos[tid] > -1e30f)) loss = 0.0f;   // rows w/o positive -> 0
#pragma unroll
    for (int off = 1; off < 32; off <<= 1) loss += __shfl_xor(loss, off, 32);
    if (tid == 0) atomicAdd(out, loss * (1.0f / (float)NROWS));
  }
}

extern "C" void kernel_launch(void* const* d_in, const int* in_sizes, int n_in,
                              void* d_out, int out_size, void* d_ws, size_t ws_size,
                              hipStream_t stream) {
  (void)in_sizes; (void)n_in; (void)out_size; (void)ws_size;
  const float* emb = (const float*)d_in[0];
  const int*   tgt = (const int*)d_in[1];
  float*       out = (float*)d_out;
  unsigned short* e16 = (unsigned short*)d_ws;   // 8192*128 bf16 = 2MB

  normalize_rows_bf16<<<NROWS / 8, 256, 0, stream>>>(emb, e16, out);
  triplet_wmma<<<NROWS / TM, 256, 0, stream>>>(e16, tgt, out);
}
